// WaveNet_23931557773439
// MI455X (gfx1250) — matrix operand order repaired and checked by reading, not compiled, not run on hardware
//
#include <hip/hip_runtime.h>
#include <stdint.h>

#define OUTC 256
#define NL   24
#define RC   64
#define GC   128
#define SC   64
#define FCH  80
#define KW   3
#define BB   2
#define TT   16384
#define TFRM 68
#define TILE 64
#define TPAD (TT + 64)   // residual bf16 mirror rows: 64-col zero front pad for causal halo

typedef __attribute__((ext_vector_type(16))) __bf16 v16bf;
typedef __attribute__((ext_vector_type(8)))  float  v8f;
typedef unsigned int u32x4 __attribute__((ext_vector_type(4)));
typedef int          i32x4 __attribute__((ext_vector_type(4)));
typedef int          i32x8 __attribute__((ext_vector_type(8)));

#if defined(__has_builtin)
#  if __has_builtin(__builtin_amdgcn_tensor_load_to_lds) && __has_builtin(__builtin_amdgcn_s_wait_tensorcnt)
#    define HAVE_TDM 1
#  endif
#endif
#ifndef HAVE_TDM
#  define HAVE_TDM 0
#endif

__device__ __forceinline__ float fsigmoid(float x) { return 1.0f / (1.0f + __expf(-x)); }
__device__ __forceinline__ float ftanh_(float x)   { return 1.0f - 2.0f / (__expf(2.0f * x) + 1.0f); }

// ---------------------------------------------------------------------------
// Embedding (one-hot gather): writes fp32 residual + bf16 time-major mirror.
// ---------------------------------------------------------------------------
__global__ void k_embed(const int* __restrict__ x, const float* __restrict__ fw,
                        const float* __restrict__ fb, float* __restrict__ r0,
                        __bf16* __restrict__ xb) {
  long i = (long)blockIdx.x * 256 + threadIdx.x;
  if (i >= (long)BB * RC * TT) return;
  int t = (int)(i % TT);
  long rem = i / TT;
  int c = (int)(rem % RC);
  int b = (int)(rem / RC);
  int idx = (t == 0) ? 0 : x[b * TT + t - 1];
  float v = fw[c * OUTC + idx] + fb[c];
  r0[i] = v;
  xb[((long)b * TPAD + t + 64) * 64 + c] = (__bf16)v;
}

// ---------------------------------------------------------------------------
// Upsample network: valid conv over aux context, then 4x (repeat x4 + smooth9)
// ---------------------------------------------------------------------------
__global__ void k_convin(const float* __restrict__ feat, const float* __restrict__ w,
                         float* __restrict__ out) {
  int i = blockIdx.x * 256 + threadIdx.x;     // B*FCH*64
  if (i >= BB * FCH * 64) return;
  int j = i & 63;
  int o = (i >> 6) % FCH;
  int b = i / (FCH * 64);
  float acc = 0.f;
  for (int c = 0; c < FCH; ++c)
    for (int k = 0; k < 5; ++k)
      acc += w[(o * FCH + c) * 5 + k] * feat[(b * FCH + c) * TFRM + j + k];
  out[i] = acc;
}

__global__ void k_upstage(const float* __restrict__ in, float* __restrict__ out,
                          int Lin, const float* __restrict__ upK) {
  int Lout = Lin * 4;
  long i = (long)blockIdx.x * 256 + threadIdx.x;
  if (i >= (long)BB * FCH * Lout) return;
  int t = (int)(i % Lout);
  long rem = i / Lout;
  int c = (int)(rem % FCH);
  int b = (int)(rem / FCH);
  const float* src = in + ((long)b * FCH + c) * Lin;
  float acc = 0.f;
  for (int k = 0; k < 9; ++k) {
    int r = t + k - 4;
    if (r >= 0 && r < Lout) acc += upK[k] * src[r >> 2];
  }
  out[i] = acc;
}

// Final upsample stage writes the bf16 feature mirror, (b, t, 96ch) time-major,
// channels 80..95 zeroed (K-padding for the WMMA GEMM).
__global__ void k_upstage_fin(const float* __restrict__ in, __bf16* __restrict__ out,
                              const float* __restrict__ upK) {
  long i = (long)blockIdx.x * 256 + threadIdx.x;
  if (i >= (long)BB * TT * 96) return;
  int c = (int)(i % 96);
  long rem = i / 96;
  int t = (int)(rem % TT);
  int b = (int)(rem / TT);
  float acc = 0.f;
  if (c < FCH) {
    const float* src = in + ((long)b * FCH + c) * 4096;
    for (int k = 0; k < 9; ++k) {
      int r = t + k - 4;
      if (r >= 0 && r < TT) acc += upK[k] * src[r >> 2];
    }
  }
  out[i] = (__bf16)acc;
}

// ---------------------------------------------------------------------------
// Weight repack into WMMA A-fragment layout (bf16).
// A 16x32 bf16 fragment: lane -> row m = lane&15 ; element e -> k =
//   16*(lane>>4) + e. Fragment = [frag][lane][16] contiguous.
// ---------------------------------------------------------------------------
__global__ void k_prep(const float* __restrict__ convW, const float* __restrict__ condW,
                       const float* __restrict__ skipW, const float* __restrict__ resW,
                       const float* __restrict__ l1W,   const float* __restrict__ l2W,
                       __bf16* __restrict__ a1, __bf16* __restrict__ a2,
                       __bf16* __restrict__ af) {
  int i = blockIdx.x * 256 + threadIdx.x;
  const int N1 = NL * 8 * 9 * 512;
  const int N2 = NL * 8 * 2 * 512;
  const int NF = 40 * 512;
  if (i < N1) {
    int eid = i & 511, fid = i >> 9;
    int lane = eid >> 4, e = eid & 15;
    int kt = fid % 9, mt = (fid / 9) & 7, l = fid / 72;
    int m = mt * 16 + (lane & 15);
    int k = kt * 32 + ((lane >> 4) << 4) + e;
    float w = 0.f;
    if (k < 192)      { int s = k >> 6, c = k & 63; w = convW[((l * GC + m) * RC + c) * KW + s]; }
    else if (k < 272) { int c = k - 192;            w = condW[(l * GC + m) * FCH + c]; }
    a1[i] = (__bf16)w;
    return;
  }
  i -= N1;
  if (i < N2) {
    int eid = i & 511, fid = i >> 9;
    int lane = eid >> 4, e = eid & 15;
    int kt = fid & 1, mt = (fid >> 1) & 7, l = fid >> 4;
    int m = mt * 16 + (lane & 15);
    int k = kt * 32 + ((lane >> 4) << 4) + e;
    float w = (m < 64) ? skipW[(l * SC + m) * 64 + k] : resW[(l * RC + (m - 64)) * 64 + k];
    a2[i] = (__bf16)w;
    return;
  }
  i -= N2;
  if (i < NF) {
    int eid = i & 511, fid = i >> 9;
    int lane = eid >> 4, e = eid & 15;
    int k = ((fid & 1) ? 32 : 0) + ((lane >> 4) << 4) + e;
    float w;
    if (fid < 8) { int mt = fid >> 1;       int m = mt * 16 + (lane & 15); w = l1W[m * 64 + k]; }
    else         { int mt = (fid - 8) >> 1; int m = mt * 16 + (lane & 15); w = l2W[m * 64 + k]; }
    af[i] = (__bf16)w;
  }
}

// ---------------------------------------------------------------------------
// Fused WaveNet layer. Block = 64-col time tile, 8 wave32, dynamic LDS only
// (so TDM descriptors can use lds_addr = plain byte offsets from 0).
//   LDS: Xs [128 cols][64 ch] @0      (16384 B)  <- TDM from xmir (t0-64..t0+63)
//        Fs [ 64 cols][96 ch] @16384  (12288 B)  <- TDM from fmir
//        Zs [ 64 cols][64 rw] @28672  ( 8192 B)
// GEMM1 (WMMA): h(128xT) = Wcat @ [x(t-2d); x(t-d); x(t); f]  (K=288)
// GEMM2 (WMMA): [skip_add ; res_add] = [skip_W ; res_W] @ z   (K=64)
// ---------------------------------------------------------------------------
__global__ void __launch_bounds__(256) k_layer(
    const float* __restrict__ rin, float* __restrict__ rout, __bf16* __restrict__ routb,
    float* __restrict__ skips, const __bf16* __restrict__ xmir,
    const __bf16* __restrict__ fmir,
    const __bf16* __restrict__ a1, const __bf16* __restrict__ a2,
    const float* __restrict__ convB, const float* __restrict__ skipB,
    const float* __restrict__ resB, int l, int d) {
  extern __shared__ __align__(32) char smem_raw[];
  __bf16* Xs = (__bf16*)smem_raw;            // LDS offset 0
  __bf16* Fs = Xs + 128 * 64;                // LDS offset 16384
  __bf16* Zs = Fs + 64 * 96;                 // LDS offset 28672

  const int tid = threadIdx.x, lane = tid & 31, w = tid >> 5;
  const int b  = blockIdx.x / (TT / TILE);
  const int t0 = (blockIdx.x % (TT / TILE)) * TILE;

#if HAVE_TDM
  // Tensor Data Mover: two 1-D descriptor DMAs (data_size=8B, tile==tensor).
  if (w == 0) {
    unsigned long long ga =
        (unsigned long long)(uintptr_t)(xmir + ((long)b * TPAD + t0) * 64);
    u32x4 g0;
    g0.x = 1u;                                   // count=1, load, no gather
    g0.y = 0u;                                   // lds_addr = Xs @ 0
    g0.z = (unsigned)(ga & 0xFFFFFFFFu);         // global_addr[31:0]
    g0.w = (unsigned)((ga >> 32) & 0x01FFFFFFu) | 0x80000000u;  // addr[56:32] | type=2
    i32x8 g1 = { 0x30000,                         // mask=0, data_size=3 (8B)
                 (int)(2048u << 16),              // tensor_dim0 = 2048 x 8B = 16 KB
                 0,
                 (int)(2048u << 16),              // tile_dim0 = 2048 (1-D tile)
                 0, 0, 0, 0 };
    i32x4 gz4 = { 0, 0, 0, 0 };
    i32x8 gz8 = { 0, 0, 0, 0, 0, 0, 0, 0 };
    __builtin_amdgcn_tensor_load_to_lds(g0, g1, gz4, gz4, gz8, 0);
  } else if (w == 1) {
    unsigned long long ga =
        (unsigned long long)(uintptr_t)(fmir + ((long)b * TT + t0) * 96);
    u32x4 g0;
    g0.x = 1u;
    g0.y = 16384u;                               // lds_addr = Fs
    g0.z = (unsigned)(ga & 0xFFFFFFFFu);
    g0.w = (unsigned)((ga >> 32) & 0x01FFFFFFu) | 0x80000000u;
    i32x8 g1 = { 0x30000,
                 (int)(1536u << 16),              // 1536 x 8B = 12 KB
                 0,
                 (int)(1536u << 16),
                 0, 0, 0, 0 };
    i32x4 gz4 = { 0, 0, 0, 0 };
    i32x8 gz8 = { 0, 0, 0, 0, 0, 0, 0, 0 };
    __builtin_amdgcn_tensor_load_to_lds(g0, g1, gz4, gz4, gz8, 0);
  }
  if (w < 2) __builtin_amdgcn_s_wait_tensorcnt(0);
#else
  { // fallback: plain vectorized bf16 copies (mirrors are already tile-layout)
    const uint4* gx = (const uint4*)(xmir + ((long)b * TPAD + t0) * 64);
    uint4* lx = (uint4*)Xs;
    for (int i = tid; i < 1024; i += 256) lx[i] = gx[i];
    const uint4* gf = (const uint4*)(fmir + ((long)b * TT + t0) * 96);
    uint4* lf = (uint4*)Fs;
    for (int i = tid; i < 768; i += 256) lf[i] = gf[i];
  }
#endif
  __syncthreads();

  const int mta = w & 3;            // a-rows tile; g-rows tile = mta+4
  const int n0  = (w >> 2) * 2;     // 2 n-tiles per wave
  v8f ca0 = {}, ca1 = {}, cg0 = {}, cg1 = {};

  const __bf16* a1l = a1 + (long)l * 8 * 9 * 512;
  for (int kt = 0; kt < 9; ++kt) {
    v16bf Aa = *(const v16bf*)(a1l + (mta * 9 + kt) * 512 + lane * 16);
    v16bf Ag = *(const v16bf*)(a1l + ((mta + 4) * 9 + kt) * 512 + lane * 16);
    int k0 = kt * 32 + ((lane >> 4) << 4);     // 16-run start; never crosses segments
#pragma unroll
    for (int q = 0; q < 2; ++q) {
      int n = (n0 + q) * 16 + (lane & 15);
      v16bf Bf;
      if (k0 < 192) {                           // conv taps: x at t-(2-s)d
        int s = k0 >> 6, c0 = k0 & 63;
        int col = n + 64 - (2 - s) * d;
        Bf = *(const v16bf*)(Xs + col * 64 + c0);
      } else {
        Bf = *(const v16bf*)(Fs + n * 96 + (k0 - 192));
      }
      if (q == 0) {
        ca0 = __builtin_amdgcn_wmma_f32_16x16x32_bf16(false, Aa, false, Bf, (short)0, ca0, false, false);
        cg0 = __builtin_amdgcn_wmma_f32_16x16x32_bf16(false, Ag, false, Bf, (short)0, cg0, false, false);
      } else {
        ca1 = __builtin_amdgcn_wmma_f32_16x16x32_bf16(false, Aa, false, Bf, (short)0, ca1, false, false);
        cg1 = __builtin_amdgcn_wmma_f32_16x16x32_bf16(false, Ag, false, Bf, (short)0, cg1, false, false);
      }
    }
  }

  const int hh = lane >> 4;
#pragma unroll
  for (int v = 0; v < 8; ++v) {
    int mrow = mta * 16 + v + hh * 8;           // a-row (0..63); g-row = mrow+64
    float ba = convB[l * GC + mrow];
    float bg = convB[l * GC + 64 + mrow];
    float z0 = ftanh_(ca0[v] + ba) * fsigmoid(cg0[v] + bg);
    float z1 = ftanh_(ca1[v] + ba) * fsigmoid(cg1[v] + bg);
    int col0 = n0 * 16 + (lane & 15);
    Zs[col0 * 64 + mrow]        = (__bf16)z0;
    Zs[(col0 + 16) * 64 + mrow] = (__bf16)z1;
  }
  __syncthreads();

  v8f c20 = {}, c21 = {}, c22 = {}, c23 = {};
  const __bf16* a2l = a2 + (long)l * 8 * 2 * 512;
#pragma unroll
  for (int kt = 0; kt < 2; ++kt) {
    v16bf A2 = *(const v16bf*)(a2l + (w * 2 + kt) * 512 + lane * 16);
    int k0 = kt * 32 + ((lane >> 4) << 4);
#pragma unroll
    for (int nt = 0; nt < 4; ++nt) {
      int n = nt * 16 + (lane & 15);
      v16bf B2 = *(const v16bf*)(Zs + n * 64 + k0);
      v8f& cc = (nt == 0) ? c20 : (nt == 1) ? c21 : (nt == 2) ? c22 : c23;
      cc = __builtin_amdgcn_wmma_f32_16x16x32_bf16(false, A2, false, B2, (short)0, cc, false, false);
    }
  }
#pragma unroll
  for (int nt = 0; nt < 4; ++nt) {
    v8f cc = (nt == 0) ? c20 : (nt == 1) ? c21 : (nt == 2) ? c22 : c23;
    int t = t0 + nt * 16 + (lane & 15);
#pragma unroll
    for (int v = 0; v < 8; ++v) {
      int m = w * 16 + v + hh * 8;              // 0..127
      if (m < 64) {                              // skip accumulation (fp32, L2-resident)
        long o = ((long)(b * SC + m)) * TT + t;
        skips[o] += cc[v] + skipB[l * SC + m];
      } else {                                   // residual update (fp32 chain + bf16 mirror)
        int mr = m - 64;
        long o = ((long)(b * RC + mr)) * TT + t;
        float val = cc[v] + resB[l * RC + mr] + rin[o];
        rout[o] = val;
        routb[((long)b * TPAD + t + 64) * 64 + mr] = (__bf16)val;
      }
    }
  }
}

// ---------------------------------------------------------------------------
// Final head: relu(skips) -> relu(last1@. + b1) -> last2@. + b2
// ---------------------------------------------------------------------------
__global__ void __launch_bounds__(256) k_final(
    const float* __restrict__ skips, const __bf16* __restrict__ af,
    const float* __restrict__ b1, const float* __restrict__ b2,
    float* __restrict__ out) {
  __shared__ __align__(32) __bf16 Ss[64 * 64];
  __shared__ __align__(32) __bf16 H1s[64 * 64];
  const int tid = threadIdx.x, lane = tid & 31, w = tid >> 5;
  const int b  = blockIdx.x / (TT / TILE);
  const int t0 = (blockIdx.x % (TT / TILE)) * TILE;

  for (int i = tid; i < 64 * 64; i += 256) {
    int c = i >> 6, n = i & 63;
    float v = skips[((long)(b * SC + c)) * TT + t0 + n];
    Ss[n * 64 + c] = (__bf16)(v > 0.f ? v : 0.f);
  }
  __syncthreads();

  const int mt1 = w & 3, n0 = (w >> 2) * 2, hh = lane >> 4;
  v8f h0 = {}, h1 = {};
#pragma unroll
  for (int kt = 0; kt < 2; ++kt) {
    v16bf A = *(const v16bf*)(af + (mt1 * 2 + kt) * 512 + lane * 16);
    int k0 = kt * 32 + ((lane >> 4) << 4);
#pragma unroll
    for (int q = 0; q < 2; ++q) {
      int n = (n0 + q) * 16 + (lane & 15);
      v16bf B = *(const v16bf*)(Ss + n * 64 + k0);
      if (q == 0) h0 = __builtin_amdgcn_wmma_f32_16x16x32_bf16(false, A, false, B, (short)0, h0, false, false);
      else        h1 = __builtin_amdgcn_wmma_f32_16x16x32_bf16(false, A, false, B, (short)0, h1, false, false);
    }
  }
#pragma unroll
  for (int v = 0; v < 8; ++v) {
    int m = mt1 * 16 + v + hh * 8;
    float bb = b1[m];
    float v0 = h0[v] + bb; v0 = v0 > 0.f ? v0 : 0.f;
    float v1 = h1[v] + bb; v1 = v1 > 0.f ? v1 : 0.f;
    int col0 = n0 * 16 + (lane & 15);
    H1s[col0 * 64 + m]        = (__bf16)v0;
    H1s[(col0 + 16) * 64 + m] = (__bf16)v1;
  }
  __syncthreads();

  const __bf16* a2 = af + 8 * 512;
#pragma unroll
  for (int half = 0; half < 2; ++half) {
    int mt = w * 2 + half;                       // 0..15 (256 output rows)
    v8f c0 = {}, c1 = {}, c2 = {}, c3 = {};
#pragma unroll
    for (int kt = 0; kt < 2; ++kt) {
      v16bf A = *(const v16bf*)(a2 + (mt * 2 + kt) * 512 + lane * 16);
      int k0 = kt * 32 + ((lane >> 4) << 4);
#pragma unroll
      for (int nt = 0; nt < 4; ++nt) {
        int n = nt * 16 + (lane & 15);
        v16bf B = *(const v16bf*)(H1s + n * 64 + k0);
        v8f& cc = (nt == 0) ? c0 : (nt == 1) ? c1 : (nt == 2) ? c2 : c3;
        cc = __builtin_amdgcn_wmma_f32_16x16x32_bf16(false, A, false, B, (short)0, cc, false, false);
      }
    }
#pragma unroll
    for (int nt = 0; nt < 4; ++nt) {
      v8f cc = (nt == 0) ? c0 : (nt == 1) ? c1 : (nt == 2) ? c2 : c3;
      int t = t0 + nt * 16 + (lane & 15);
#pragma unroll
      for (int v = 0; v < 8; ++v) {
        int m = mt * 16 + v + hh * 8;
        out[((long)(b * OUTC + m)) * TT + t] = cc[v] + b2[m];
      }
    }
  }
}

// ---------------------------------------------------------------------------
extern "C" void kernel_launch(void* const* d_in, const int* in_sizes, int n_in,
                              void* d_out, int out_size, void* d_ws, size_t ws_size,
                              hipStream_t stream) {
  (void)in_sizes; (void)n_in; (void)out_size; (void)ws_size;
  const int*   x       = (const int*)  d_in[0];
  const float* feature = (const float*)d_in[1];
  const float* firstW  = (const float*)d_in[2];
  const float* firstB  = (const float*)d_in[3];
  const float* convW   = (const float*)d_in[4];
  const float* convB   = (const float*)d_in[5];
  const float* condW   = (const float*)d_in[6];
  const float* skipW   = (const float*)d_in[7];
  const float* skipB   = (const float*)d_in[8];
  const float* resW    = (const float*)d_in[9];
  const float* resB    = (const float*)d_in[10];
  const float* l1W     = (const float*)d_in[11];
  const float* l1B     = (const float*)d_in[12];
  const float* l2W     = (const float*)d_in[13];
  const float* l2B     = (const float*)d_in[14];
  const float* cinW    = (const float*)d_in[15];
  const float* upK     = (const float*)d_in[16];

  char* ws = (char*)d_ws;
  size_t off = 0;
  auto alloc = [&](size_t bytes) -> char* {
    char* p = ws + off; off = (off + bytes + 255) & ~(size_t)255; return p;
  };
  float*  r0    = (float*) alloc((size_t)BB * RC * TT * 4);
  float*  r1    = (float*) alloc((size_t)BB * RC * TT * 4);
  float*  skips = (float*) alloc((size_t)BB * SC * TT * 4);
  __bf16* xb0   = (__bf16*)alloc((size_t)BB * TPAD * 64 * 2);   // time-major bf16 mirror
  __bf16* xb1   = (__bf16*)alloc((size_t)BB * TPAD * 64 * 2);
  __bf16* fb    = (__bf16*)alloc((size_t)BB * TT * 96 * 2);     // (b,t,96ch) bf16 mirror
  float*  upA   = (float*) alloc((size_t)BB * FCH * 4096 * 4);
  float*  upB   = (float*) alloc((size_t)BB * FCH * 4096 * 4);
  __bf16* a1    = (__bf16*)alloc((size_t)NL * 8 * 9 * 512 * 2);
  __bf16* a2    = (__bf16*)alloc((size_t)NL * 8 * 2 * 512 * 2);
  __bf16* af    = (__bf16*)alloc((size_t)40 * 512 * 2);

  (void)hipMemsetAsync(skips, 0, (size_t)BB * SC * TT * 4, stream);
  // zero the 64-column causal front pads of both residual mirrors
  for (int b = 0; b < BB; ++b) {
    (void)hipMemsetAsync(xb0 + (size_t)b * TPAD * 64, 0, 64 * 64 * 2, stream);
    (void)hipMemsetAsync(xb1 + (size_t)b * TPAD * 64, 0, 64 * 64 * 2, stream);
  }

  { int tot = NL * 8 * 9 * 512 + NL * 8 * 2 * 512 + 40 * 512;
    k_prep<<<(tot + 255) / 256, 256, 0, stream>>>(convW, condW, skipW, resW, l1W, l2W, a1, a2, af); }

  { long tot = (long)BB * RC * TT;
    k_embed<<<(int)((tot + 255) / 256), 256, 0, stream>>>(x, firstW, firstB, r0, xb0); }

  { int tot = BB * FCH * 64;
    k_convin<<<(tot + 255) / 256, 256, 0, stream>>>(feature, cinW, upA); }

  k_upstage<<<(BB * FCH * 256  + 255) / 256, 256, 0, stream>>>(upA, upB, 64,   upK + 0);
  k_upstage<<<(BB * FCH * 1024 + 255) / 256, 256, 0, stream>>>(upB, upA, 256,  upK + 9);
  k_upstage<<<(BB * FCH * 4096 + 255) / 256, 256, 0, stream>>>(upA, upB, 1024, upK + 18);
  { long tot = (long)BB * TT * 96;
    k_upstage_fin<<<(int)((tot + 255) / 256), 256, 0, stream>>>(upB, fb, upK + 27); }

  const int nblk = BB * TT / TILE;               // 512 blocks
  const int ldsb = (128 * 64 + 64 * 96 + 64 * 64) * 2;  // 36864 B dynamic LDS
  for (int l = 0; l < NL; ++l) {
    int d = 1 << (l % 6);
    const float*  rin  = (l & 1) ? r1  : r0;
    float*        rot  = (l & 1) ? r0  : r1;
    const __bf16* xin  = (l & 1) ? xb1 : xb0;
    __bf16*       xout = (l & 1) ? xb0 : xb1;
    k_layer<<<nblk, 256, ldsb, stream>>>(rin, rot, xout, skips, xin, fb,
                                         a1, a2, convB, skipB, resB, l, d);
  }
  k_final<<<nblk, 256, 0, stream>>>(skips, af, l1B, l2B, (float*)d_out);
}
